// NeuralKnapsackSolver_34359738368346
// MI455X (gfx1250) — matrix operand order repaired and checked by reading, not compiled, not run on hardware
//
#include <hip/hip_runtime.h>
#include <math.h>

// Problem sizes (match reference)
#define BB 256
#define TT 512
#define DD 128
#define HH 256
#define GG 768   // 3*H

typedef __attribute__((ext_vector_type(16))) __bf16          v16bf;
typedef __attribute__((ext_vector_type(16))) unsigned short  v16us;
typedef __attribute__((ext_vector_type(8)))  unsigned short  v8us;
typedef __attribute__((ext_vector_type(8)))  float           v8f;

__device__ __forceinline__ unsigned short f2bf(float f) {
    unsigned int u = __float_as_uint(f);
    u += 0x7FFFu + ((u >> 16) & 1u);          // round-to-nearest-even
    return (unsigned short)(u >> 16);
}
__device__ __forceinline__ float bf2f(unsigned short h) {
    return __uint_as_float(((unsigned int)h) << 16);
}
__device__ __forceinline__ float sigf(float x) { return 1.0f / (1.0f + __expf(-x)); }

#define WMMA_BF16(a, b, c) \
    __builtin_amdgcn_wmma_f32_16x16x32_bf16(false, (a), false, (b), (short)0, (c), false, false)

// ---------------------------------------------------------------------------
// A-fragment: 16x32 bf16 tile, row-major source (LDS), lda in elements.
// Layout used here: lane m=l&15, kg=l>>4; elems 0..15 -> K = 16*kg + e.
//             Two contiguous 16B runs.
// ---------------------------------------------------------------------------
__device__ __forceinline__ v16bf load_a_frag(const unsigned short* Abase, int lda, int k0) {
    int lane = threadIdx.x & 31;
    int m = lane & 15, kg = lane >> 4;
    const unsigned short* p0 = Abase + m * lda + k0 + 16 * kg;
    v8us lo = *(const v8us*)p0;
    v8us hi = *(const v8us*)(p0 + 8);
    v16us t;
#pragma unroll
    for (int e = 0; e < 8; e++) { t[e] = lo[e]; t[8 + e] = hi[e]; }
    return __builtin_bit_cast(v16bf, t);
}

// ---------------------------------------------------------------------------
// B-fragment: 32x16 tile of B = W^T, W row-major [N, K] (bf16), ldk elements.
// Layout used here: lane n=l&15, kg=l>>4; elems 0..15 -> K = 16*kg + e.
// One contiguous 32B run per lane.
// ---------------------------------------------------------------------------
__device__ __forceinline__ v16bf load_b_frag(const unsigned short* W, int ldk, int n0, int k0) {
    int lane = threadIdx.x & 31;
    int n = lane & 15, kg = lane >> 4;
    const unsigned short* p = W + (size_t)(n0 + n) * ldk + k0 + 16 * kg;
    v16us t = *(const v16us*)p;
    return __builtin_bit_cast(v16bf, t);
}

// ---------------------------------------------------------------------------
// fp32 -> bf16 weight conversion
// ---------------------------------------------------------------------------
__global__ void k_cvt(const float* __restrict__ s, unsigned short* __restrict__ d, int n) {
    int i = blockIdx.x * blockDim.x + threadIdx.x;
    if (i < n) d[i] = f2bf(s[i]);
}

// ---------------------------------------------------------------------------
// Fused zero-state GRU cell: h = (1-z)*n,  r=sig(gr+bhr), z=sig(gz+bhz),
// n=tanh(gn + r*bhn), g = A @ W^T + bi.  A: [16 x K] bf16 in LDS.
// Wave w owns matched column tiles {2w+i, 16+2w+i, 32+2w+i} so the whole
// gate nonlinearity fuses in registers.  hout must NOT alias A (ping-pong
// buffers) -> results stream straight to LDS, one barrier per cell.
// ---------------------------------------------------------------------------
template <int KT>
__device__ inline void cell0_layer(const unsigned short* A, int lda,
                                   const unsigned short* W, int ldk,
                                   const float* bi, const float* bh,
                                   unsigned short* hout) {
    __syncthreads();   // previous producer of A finished writing
    int tid = threadIdx.x, wave = tid >> 5, lane = tid & 31, ln = lane & 15, kg = lane >> 4;
#pragma unroll
    for (int i = 0; i < 2; i++) {
        int ct = wave * 2 + i;   // 0..15
        v8f aR = {0, 0, 0, 0, 0, 0, 0, 0};
        v8f aZ = aR, aN = aR;
#pragma unroll 2
        for (int kt = 0; kt < KT; kt++) {
            v16bf a = load_a_frag(A, lda, kt * 32);
            aR = WMMA_BF16(a, load_b_frag(W, ldk, ct * 16, kt * 32), aR);
            aZ = WMMA_BF16(a, load_b_frag(W, ldk, (16 + ct) * 16, kt * 32), aZ);
            aN = WMMA_BF16(a, load_b_frag(W, ldk, (32 + ct) * 16, kt * 32), aN);
        }
        int c = ct * 16 + ln;
        float biR = bi[c],          bhR = bh[c];
        float biZ = bi[HH + c],     bhZ = bh[HH + c];
        float biN = bi[2 * HH + c], bhN = bh[2 * HH + c];
#pragma unroll
        for (int v = 0; v < 8; v++) {
            float r  = sigf(aR[v] + biR + bhR);
            float z  = sigf(aZ[v] + biZ + bhZ);
            float nn = tanhf(aN[v] + biN + r * bhN);
            hout[(v + 8 * kg) * HH + c] = f2bf((1.0f - z) * nn);
        }
    }
}

// ---------------------------------------------------------------------------
// Phase 1: memory GRU over T.  One workgroup per 16 batch rows, loops t.
// Register-fused gates; hidden state ping-pongs between two LDS buffers.
// ---------------------------------------------------------------------------
__global__ __launch_bounds__(256) void k_memgru(const float* __restrict__ x,
                                                const unsigned short* __restrict__ Wi,
                                                const unsigned short* __restrict__ Wh,
                                                const float* __restrict__ bi,
                                                const float* __restrict__ bh,
                                                unsigned short* __restrict__ mem16) {
    __shared__ __align__(32) unsigned short sh_x[16 * DD];       // 4KB
    __shared__ __align__(32) unsigned short sh_h[2][16 * HH];    // 2 x 8KB ping-pong
    int tid = threadIdx.x, wave = tid >> 5, lane = tid & 31, ln = lane & 15, kg = lane >> 4;
    int b0 = blockIdx.x * 16;

    for (int i = tid; i < 16 * HH; i += 256) sh_h[0][i] = 0;
    __syncthreads();

    int p = 0;
    for (int t = 0; t < TT; t++) {
        const unsigned short* hprev = sh_h[p];
        unsigned short*       hnext = sh_h[p ^ 1];
        // stage x_t tile as bf16
        for (int i = tid; i < 16 * DD; i += 256) {
            int m = i >> 7, k = i & (DD - 1);
            sh_x[i] = f2bf(x[((size_t)(b0 + m) * TT + t) * DD + k]);
        }
        __syncthreads();

#pragma unroll
        for (int i = 0; i < 2; i++) {
            int ct = wave * 2 + i;
            v8f aR = {0, 0, 0, 0, 0, 0, 0, 0};
            v8f aZ = aR, aNi = aR, aNh = aR;
            // gi = x_t @ Wi^T  (K=128): r,z accumulate; n kept separate (aNi)
#pragma unroll 2
            for (int kt = 0; kt < DD / 32; kt++) {
                v16bf a = load_a_frag(sh_x, DD, kt * 32);
                aR  = WMMA_BF16(a, load_b_frag(Wi, DD, ct * 16, kt * 32), aR);
                aZ  = WMMA_BF16(a, load_b_frag(Wi, DD, (16 + ct) * 16, kt * 32), aZ);
                aNi = WMMA_BF16(a, load_b_frag(Wi, DD, (32 + ct) * 16, kt * 32), aNi);
            }
            // gh = h @ Wh^T  (K=256): r,z continue same acc; n separate (aNh)
#pragma unroll 2
            for (int kt = 0; kt < HH / 32; kt++) {
                v16bf a = load_a_frag(hprev, HH, kt * 32);
                aR  = WMMA_BF16(a, load_b_frag(Wh, HH, ct * 16, kt * 32), aR);
                aZ  = WMMA_BF16(a, load_b_frag(Wh, HH, (16 + ct) * 16, kt * 32), aZ);
                aNh = WMMA_BF16(a, load_b_frag(Wh, HH, (32 + ct) * 16, kt * 32), aNh);
            }
            int c = ct * 16 + ln;
            float biR = bi[c],          bhR = bh[c];
            float biZ = bi[HH + c],     bhZ = bh[HH + c];
            float biN = bi[2 * HH + c], bhN = bh[2 * HH + c];
#pragma unroll
            for (int v = 0; v < 8; v++) {
                int m = v + 8 * kg;
                float r  = sigf(aR[v] + biR + bhR);
                float z  = sigf(aZ[v] + biZ + bhZ);
                float nn = tanhf(aNi[v] + biN + r * (aNh[v] + bhN));
                float ho = bf2f(hprev[m * HH + c]);
                unsigned short us = f2bf((1.0f - z) * nn + z * ho);
                hnext[m * HH + c] = us;
                mem16[((size_t)(b0 + m) * TT + t) * HH + c] = us;
            }
        }
        __syncthreads();   // hnext complete before next step's gemm reads it
        p ^= 1;
    }
}

// ---------------------------------------------------------------------------
// Phase 2: decoder scan.  Attention over mem (bf16, L2-resident) + 5 fused
// zero-state GRU cells + sigmoid output.  Cell chain ping-pongs hb0/hb1;
// carried knapsack state is always hb0.
// ---------------------------------------------------------------------------
__global__ __launch_bounds__(256) void k_decoder(
    const float* __restrict__ x, const unsigned short* __restrict__ mem16,
    const unsigned short* __restrict__ Wd0, const float* __restrict__ bid0, const float* __restrict__ bhd0,
    const unsigned short* __restrict__ Wd1, const float* __restrict__ bid1, const float* __restrict__ bhd1,
    const unsigned short* __restrict__ Wd2, const float* __restrict__ bid2, const float* __restrict__ bhd2,
    const unsigned short* __restrict__ Wk0, const float* __restrict__ bik0, const float* __restrict__ bhk0,
    const unsigned short* __restrict__ Wk1, const float* __restrict__ bik1, const float* __restrict__ bhk1,
    const float* __restrict__ Wo, const float* __restrict__ bo, float* __restrict__ out) {
    __shared__ __align__(32) float          sh_sc[16 * TT];        // 32KB softmax buf
    __shared__ __align__(32) unsigned short sh_d[16 * (DD + HH)];  // 12KB [x_t | ctx]
    __shared__ __align__(32) unsigned short sh_hb0[16 * HH];       // 8KB (also ksack)
    __shared__ __align__(32) unsigned short sh_hb1[16 * HH];       // 8KB
    int tid = threadIdx.x, wave = tid >> 5, lane = tid & 31;
    int b0 = blockIdx.x * 16;

    for (int i = tid; i < 16 * HH; i += 256) sh_hb0[i] = 0;   // ksack0 = 0
    __syncthreads();

    for (int t = 0; t < TT; t++) {
        // stage x_t into d[:, 0:128]
        for (int i = tid; i < 16 * DD; i += 256) {
            int m = i >> 7, k = i & (DD - 1);
            sh_d[m * (DD + HH) + k] = f2bf(x[((size_t)(b0 + m) * TT + t) * DD + k]);
        }
        // attention scores: scores[row, t'] = mem[row,t',:] . ksack[row,:]
        for (int idx = tid; idx < 16 * TT; idx += 256) {
            int row = idx >> 9, tp = idx & (TT - 1);
            const uint4* mp = (const uint4*)(mem16 + ((size_t)(b0 + row) * TT + tp) * HH);
            const unsigned short* kr = sh_hb0 + row * HH;
            float s = 0.0f;
#pragma unroll 4
            for (int cc = 0; cc < HH / 8; cc++) {
                uint4 q = mp[cc];
                int h0 = cc * 8;
                s += __uint_as_float(q.x << 16)          * bf2f(kr[h0 + 0]);
                s += __uint_as_float(q.x & 0xFFFF0000u)  * bf2f(kr[h0 + 1]);
                s += __uint_as_float(q.y << 16)          * bf2f(kr[h0 + 2]);
                s += __uint_as_float(q.y & 0xFFFF0000u)  * bf2f(kr[h0 + 3]);
                s += __uint_as_float(q.z << 16)          * bf2f(kr[h0 + 4]);
                s += __uint_as_float(q.z & 0xFFFF0000u)  * bf2f(kr[h0 + 5]);
                s += __uint_as_float(q.w << 16)          * bf2f(kr[h0 + 6]);
                s += __uint_as_float(q.w & 0xFFFF0000u)  * bf2f(kr[h0 + 7]);
            }
            sh_sc[idx] = s;
        }
        __syncthreads();
        // softmax per row (wave32 shfl reductions; wave w -> rows w, w+8)
        for (int rr = wave; rr < 16; rr += 8) {
            float mx = -3.0e38f;
            for (int tp = lane; tp < TT; tp += 32) mx = fmaxf(mx, sh_sc[rr * TT + tp]);
#pragma unroll
            for (int o = 16; o > 0; o >>= 1) mx = fmaxf(mx, __shfl_xor(mx, o, 32));
            float sm = 0.0f;
            for (int tp = lane; tp < TT; tp += 32) {
                float e = __expf(sh_sc[rr * TT + tp] - mx);
                sh_sc[rr * TT + tp] = e;
                sm += e;
            }
#pragma unroll
            for (int o = 16; o > 0; o >>= 1) sm += __shfl_xor(sm, o, 32);
            float inv = 1.0f / sm;
            for (int tp = lane; tp < TT; tp += 32) sh_sc[rr * TT + tp] *= inv;
        }
        __syncthreads();
        // ctx[row,h] = sum_t w[t]*mem[row,t,h]  -> d[:, 128:384]
        for (int idx = tid; idx < 16 * HH; idx += 256) {
            int row = idx >> 8, h = idx & (HH - 1);
            const unsigned short* mcol = mem16 + ((size_t)(b0 + row) * TT) * HH + h;
            float s = 0.0f;
            for (int tp = 0; tp < TT; tp++) s += sh_sc[row * TT + tp] * bf2f(mcol[(size_t)tp * HH]);
            sh_d[row * (DD + HH) + DD + h] = f2bf(s);
        }
        // 5 fused zero-state GRU cells (WMMA); each starts with a barrier
        cell0_layer<(DD + HH) / 32>(sh_d, DD + HH, Wd0, DD + HH, bid0, bhd0, sh_hb0 == sh_hb0 ? sh_hb1 : sh_hb1); // -> hb1
        cell0_layer<HH / 32>(sh_hb1, HH, Wd1, HH, bid1, bhd1, sh_hb0);   // hb1 -> hb0
        cell0_layer<HH / 32>(sh_hb0, HH, Wd2, HH, bid2, bhd2, sh_hb1);   // hb0 -> hb1
        cell0_layer<HH / 32>(sh_hb1, HH, Wk0, HH, bik0, bhk0, sh_hb0);   // hb1 -> hb0
        cell0_layer<HH / 32>(sh_hb0, HH, Wk1, HH, bik1, bhk1, sh_hb1);   // hb0 -> hb1
        __syncthreads();
        // copy final k into hb0 (carried ksack) while computing output
        for (int i = tid; i < 16 * HH; i += 256) sh_hb0[i] = sh_hb1[i];
        // output: sigmoid(k . Wo + bo)
        for (int rr = wave; rr < 16; rr += 8) {
            float s = 0.0f;
            for (int h = lane; h < HH; h += 32) s += bf2f(sh_hb1[rr * HH + h]) * Wo[h];
#pragma unroll
            for (int o = 16; o > 0; o >>= 1) s += __shfl_xor(s, o, 32);
            if (lane == 0) out[(size_t)(b0 + rr) * TT + t] = sigf(s + bo[0]);
        }
        __syncthreads();  // hb0 (ksack) stable before next step's scores
    }
}

// ---------------------------------------------------------------------------
extern "C" void kernel_launch(void* const* d_in, const int* in_sizes, int n_in,
                              void* d_out, int out_size, void* d_ws, size_t ws_size,
                              hipStream_t stream) {
    (void)in_sizes; (void)n_in; (void)out_size; (void)ws_size;
    const float* x     = (const float*)d_in[0];
    const float* Wm_i  = (const float*)d_in[1];
    const float* Wm_h  = (const float*)d_in[2];
    const float* bm_i  = (const float*)d_in[3];
    const float* bm_h  = (const float*)d_in[4];
    const float* Wd_i0 = (const float*)d_in[5];
    const float* bd_i0 = (const float*)d_in[6];
    const float* bd_h0 = (const float*)d_in[7];
    const float* Wd_i1 = (const float*)d_in[8];
    const float* bd_i1 = (const float*)d_in[9];
    const float* bd_h1 = (const float*)d_in[10];
    const float* Wd_i2 = (const float*)d_in[11];
    const float* bd_i2 = (const float*)d_in[12];
    const float* bd_h2 = (const float*)d_in[13];
    const float* Wk_i0 = (const float*)d_in[14];
    const float* bk_i0 = (const float*)d_in[15];
    const float* bk_h0 = (const float*)d_in[16];
    const float* Wk_i1 = (const float*)d_in[17];
    const float* bk_i1 = (const float*)d_in[18];
    const float* bk_h1 = (const float*)d_in[19];
    const float* Wo    = (const float*)d_in[20];
    const float* bo    = (const float*)d_in[21];

    char* ws = (char*)d_ws;
    size_t off = 0;
    auto take = [&](size_t elems) -> unsigned short* {
        unsigned short* p = (unsigned short*)(ws + off);
        off = (off + elems * 2 + 255) & ~(size_t)255;
        return p;
    };
    unsigned short* wWmi  = take((size_t)GG * DD);
    unsigned short* wWmh  = take((size_t)GG * HH);
    unsigned short* wWd0  = take((size_t)GG * (DD + HH));
    unsigned short* wWd1  = take((size_t)GG * HH);
    unsigned short* wWd2  = take((size_t)GG * HH);
    unsigned short* wWk0  = take((size_t)GG * HH);
    unsigned short* wWk1  = take((size_t)GG * HH);
    unsigned short* mem16 = take((size_t)BB * TT * HH);   // 64 MB bf16 (L2-resident)

    auto cvt = [&](const float* s, unsigned short* d, int n) {
        k_cvt<<<dim3((n + 255) / 256), dim3(256), 0, stream>>>(s, d, n);
    };
    cvt(Wm_i, wWmi, GG * DD);
    cvt(Wm_h, wWmh, GG * HH);
    cvt(Wd_i0, wWd0, GG * (DD + HH));
    cvt(Wd_i1, wWd1, GG * HH);
    cvt(Wd_i2, wWd2, GG * HH);
    cvt(Wk_i0, wWk0, GG * HH);
    cvt(Wk_i1, wWk1, GG * HH);

    k_memgru<<<dim3(BB / 16), dim3(256), 0, stream>>>(x, wWmi, wWmh, bm_i, bm_h, mem16);
    k_decoder<<<dim3(BB / 16), dim3(256), 0, stream>>>(
        x, mem16,
        wWd0, bd_i0, bd_h0, wWd1, bd_i1, bd_h1, wWd2, bd_i2, bd_h2,
        wWk0, bk_i0, bk_h0, wWk1, bk_i1, bk_h1,
        Wo, bo, (float*)d_out);
}
